// LocalAttentionModule_74491912782022
// MI455X (gfx1250) — matrix operand order repaired and checked by reading, not compiled, not run on hardware
//
#include <hip/hip_runtime.h>
#include <stdint.h>

// ---------------------------------------------------------------------------
// LocalAttentionModule for MI455X (gfx1250): bitset 2-hop mask + flash
// attention built on v_wmma_f32_16x16x32_bf16 (16x16x32, f32 accumulate).
// N=4096, D=256, H=8, HD=32 (HD == WMMA K for bf16 -> one WMMA per tile).
// All WMMA B-operands are laid out K-contiguous (weights pre-transposed,
// V pre-transposed) so every operand load is a global_load_b128.
// ---------------------------------------------------------------------------

#define N_NODES 4096
#define DMODEL  256
#define NHEADS  8
#define HDIM    32
#define MASK_W  128          // 4096 bits / 32
#define NEGBIG  (-1e9f)

typedef __attribute__((ext_vector_type(16))) __bf16 bf16x16;
typedef __attribute__((ext_vector_type(8)))  float  f32x8;

union BfPack { uint4 u; __bf16 h[8]; };

__device__ __forceinline__ f32x8 f32x8_zero() {
  f32x8 z = {0.f, 0.f, 0.f, 0.f, 0.f, 0.f, 0.f, 0.f};
  return z;
}

__device__ __forceinline__ f32x8 wmma_bf16(bf16x16 a, bf16x16 b, f32x8 c) {
  // D = A(16x32 bf16) * B(32x16 bf16) + C(16x16 f32)
  return __builtin_amdgcn_wmma_f32_16x16x32_bf16(
      /*neg_a=*/false, a, /*neg_b=*/false, b,
      /*c_mod=*/(short)0, c, /*reuse_a=*/false, /*reuse_b=*/false);
}

// A operand 16x32: lane m=lane&15, hs=lane>>4; elems 0..7 -> K=hs*8+e,
// elems 8..15 -> K=16+hs*8+(e-8). Row-major source, both chunks contiguous.
__device__ __forceinline__ bf16x16 load_A16x32(const __bf16* base, int ld,
                                               int row0, int k0) {
  int lane = threadIdx.x & 31;
  int m = lane & 15, hs = lane >> 4;
  const __bf16* p = base + (size_t)(row0 + m) * ld + k0 + hs * 8;
  BfPack a0, a1;
  a0.u = *(const uint4*)p;
  a1.u = *(const uint4*)(p + 16);
  bf16x16 A;
#pragma unroll
  for (int i = 0; i < 8; ++i) { A[i] = a0.h[i]; A[8 + i] = a1.h[i]; }
  return A;
}

// B operand 32x16 where the per-lane K run is contiguous in memory:
// B[k][n] = base[(n0+n)*ld + k0 + k]   (lane n=lane&15, k = hs*8 + e, e>=8: 16+hs*8+(e-8))
__device__ __forceinline__ bf16x16 load_B_kcontig(const __bf16* base, int ld,
                                                  int n0, int k0) {
  int lane = threadIdx.x & 31;
  int n = lane & 15, hs = lane >> 4;
  const __bf16* p = base + (size_t)(n0 + n) * ld + k0 + hs * 8;
  BfPack b0, b1;
  b0.u = *(const uint4*)p;
  b1.u = *(const uint4*)(p + 16);
  bf16x16 B;
#pragma unroll
  for (int i = 0; i < 8; ++i) { B[i] = b0.h[i]; B[8 + i] = b1.h[i]; }
  return B;
}

// ---------------------------------------------------------------------------
// 1) adjacency bitset build
// ---------------------------------------------------------------------------
__global__ void zero_words(uint32_t* p, int n) {
  for (int i = blockIdx.x * blockDim.x + threadIdx.x; i < n;
       i += gridDim.x * blockDim.x)
    p[i] = 0u;
}

__global__ void build_adj(const int* ei, int E, uint32_t* adj) {
  int total = E + N_NODES;
  for (int i = blockIdx.x * blockDim.x + threadIdx.x; i < total;
       i += gridDim.x * blockDim.x) {
    if (i < E) {
      int s = ei[i], d = ei[E + i];
      atomicOr(&adj[(size_t)s * MASK_W + (d >> 5)], 1u << (d & 31));
      atomicOr(&adj[(size_t)d * MASK_W + (s >> 5)], 1u << (s & 31));
    } else {
      int n = i - E;
      atomicOr(&adj[(size_t)n * MASK_W + (n >> 5)], 1u << (n & 31));
    }
  }
}

// mask row i = OR over neighbors k of adj row k  (== (A@A)>0, A has self loops)
__global__ void reach2(const uint32_t* adj, uint32_t* mask) {
  __shared__ uint32_t row[MASK_W];
  __shared__ unsigned short nbrs[N_NODES];
  __shared__ int cnt;
  int i = blockIdx.x;
  int t = threadIdx.x;               // 128 threads
  if (t == 0) cnt = 0;
  __syncthreads();
  uint32_t w = adj[(size_t)i * MASK_W + t];
  row[t] = w;
  while (w) {
    int b = __ffs(w) - 1;
    w &= (w - 1);
    int pos = atomicAdd(&cnt, 1);
    nbrs[pos] = (unsigned short)(t * 32 + b);
  }
  __syncthreads();
  int nn = cnt;
  uint32_t acc = 0;
  for (int j = 0; j < nn; ++j)
    acc |= adj[(size_t)nbrs[j] * MASK_W + t];
  mask[(size_t)i * MASK_W + t] = acc;
}

// ---------------------------------------------------------------------------
// 2) f32 -> bf16 conversion (plain, and transposing for weights)
// ---------------------------------------------------------------------------
__global__ void cvt_bf16(const float* src, __bf16* dst, int n) {
  for (int i = blockIdx.x * blockDim.x + threadIdx.x; i < n;
       i += gridDim.x * blockDim.x)
    dst[i] = (__bf16)src[i];
}

// src row-major f32 [rows][cols]  ->  dst bf16 transposed [cols][rows]
__global__ void cvt_bf16_T(const float* src, __bf16* dst, int rows, int cols) {
  int total = rows * cols;
  for (int i = blockIdx.x * blockDim.x + threadIdx.x; i < total;
       i += gridDim.x * blockDim.x) {
    int r = i / cols, c = i - r * cols;
    dst[(size_t)c * rows + r] = (__bf16)src[i];
  }
}

// ---------------------------------------------------------------------------
// 3) QKV projections. One wave per 16x16 output tile, K = 256 in 8 WMMA steps.
//    Weights are pre-transposed bf16 (WT[n][k]) so B loads are b128.
//    mat 0 -> Q (row-major bf16), 1 -> K (row-major bf16), 2 -> V^T [256][4096]
// ---------------------------------------------------------------------------
__global__ void qkv_gemm(const __bf16* xb,
                         const __bf16* WqT, const __bf16* WkT, const __bf16* WvT,
                         const float* bq, const float* bk, const float* bv,
                         __bf16* Qb, __bf16* Kb, __bf16* Vt) {
  int wid = blockIdx.x * (blockDim.x >> 5) + (threadIdx.x >> 5);
  int mat = wid >> 12;               // 4096 tiles per matrix
  int t   = wid & 4095;
  int row0 = (t >> 4) * 16;
  int col0 = (t & 15) * 16;
  const __bf16* WT  = (mat == 0) ? WqT : (mat == 1) ? WkT : WvT;
  const float* bias = (mat == 0) ? bq : (mat == 1) ? bk : bv;

  f32x8 acc = f32x8_zero();
#pragma unroll
  for (int k0 = 0; k0 < DMODEL; k0 += 32) {
    bf16x16 A = load_A16x32(xb, DMODEL, row0, k0);
    bf16x16 B = load_B_kcontig(WT, DMODEL, col0, k0);  // B[k][n] = WT[n][k]
    acc = wmma_bf16(A, B, acc);
  }
  int lane = threadIdx.x & 31;
  int n = lane & 15, hs = lane >> 4;
  float bv_ = bias[col0 + n];
  __bf16* dst = (mat == 0) ? Qb : Kb;
#pragma unroll
  for (int i = 0; i < 8; ++i) {
    float v = acc[i] + bv_;
    int m = row0 + i + hs * 8;
    int g = col0 + n;
    if (mat < 2) dst[(size_t)m * DMODEL + g] = (__bf16)v;
    else         Vt[(size_t)g * N_NODES + m] = (__bf16)v;   // V transposed
  }
}

// ---------------------------------------------------------------------------
// 4) Flash attention. One wave per (head, 16-row query tile); streams keys in
//    chunks of 32 (two 16x16 score WMMAs + two PV WMMAs per chunk).
// ---------------------------------------------------------------------------
__global__ void attn_flash(const __bf16* Qb, const __bf16* Kb, const __bf16* Vt,
                           const uint32_t* mask, float* attended) {
  __shared__ __align__(16) __bf16 plds[8][16 * 32];   // probs staging per wave
  int warp = threadIdx.x >> 5;
  int wid  = blockIdx.x * 8 + warp;                   // 2048 waves total
  int h    = wid >> 8;
  int row0 = (wid & 255) * 16;
  int lane = threadIdx.x & 31;
  int nn = lane & 15, hs = lane >> 4;
  const float scale = 0.17677669529663687f;           // 1/sqrt(32)

  bf16x16 Aq = load_A16x32(Qb, DMODEL, row0, h * HDIM);

  float m_[8], l_[8];
  f32x8 acc0 = f32x8_zero(), acc1 = f32x8_zero();
#pragma unroll
  for (int i = 0; i < 8; ++i) { m_[i] = NEGBIG; l_[i] = 0.f; }

  for (int key0 = 0; key0 < N_NODES; key0 += 32) {
    // prefetch next key/value tiles (global_prefetch_b8)
    if (key0 + 32 < N_NODES) {
      __builtin_prefetch(Kb + (size_t)(key0 + 32 + lane) * DMODEL + h * HDIM, 0, 1);
      __builtin_prefetch(Vt + (size_t)(h * HDIM + lane) * N_NODES + key0 + 32, 0, 1);
    }

    // scores: Q(16x32) @ K^T(32x16) per key sub-tile
    bf16x16 Bk0 = load_B_kcontig(Kb, DMODEL, key0,      h * HDIM);
    bf16x16 Bk1 = load_B_kcontig(Kb, DMODEL, key0 + 16, h * HDIM);
    f32x8 s0 = wmma_bf16(Aq, Bk0, f32x8_zero());
    f32x8 s1 = wmma_bf16(Aq, Bk1, f32x8_zero());

    float p0[8], p1[8];
#pragma unroll
    for (int i = 0; i < 8; ++i) {
      int q = row0 + i + hs * 8;
      uint32_t w = mask[(size_t)q * MASK_W + (key0 >> 5)];   // 32 key bits
      float v0 = ((w >> nn) & 1u)        ? s0[i] * scale : NEGBIG;
      float v1 = ((w >> (nn + 16)) & 1u) ? s1[i] * scale : NEGBIG;
      float tm = fmaxf(v0, v1);
#pragma unroll
      for (int off = 1; off < 16; off <<= 1)
        tm = fmaxf(tm, __shfl_xor(tm, off, 32));
      float nm = fmaxf(m_[i], tm);
      float alpha = __expf(m_[i] - nm);                // exp(-1e9-..)->0 wipes
      float e0 = __expf(v0 - nm);
      float e1 = __expf(v1 - nm);
      float rs = e0 + e1;
#pragma unroll
      for (int off = 1; off < 16; off <<= 1)
        rs += __shfl_xor(rs, off, 32);
      l_[i] = l_[i] * alpha + rs;
      m_[i] = nm;
      acc0[i] *= alpha;
      acc1[i] *= alpha;
      p0[i] = e0;
      p1[i] = e1;
    }

    // C-layout -> A-layout via LDS (wave-private; LDS ops in-order per wave)
#pragma unroll
    for (int i = 0; i < 8; ++i) {
      int q = i + hs * 8;
      plds[warp][q * 32 + nn]      = (__bf16)p0[i];
      plds[warp][q * 32 + 16 + nn] = (__bf16)p1[i];
    }
    __builtin_amdgcn_wave_barrier();
    bf16x16 Ap;
    {
      BfPack c0, c1;
      c0.u = *(const uint4*)(&plds[warp][nn * 32 + hs * 8]);
      c1.u = *(const uint4*)(&plds[warp][nn * 32 + 16 + hs * 8]);
#pragma unroll
      for (int i = 0; i < 8; ++i) { Ap[i] = c0.h[i]; Ap[8 + i] = c1.h[i]; }
    }

    // PV: probs(16x32) @ V(32keys x 32d) -> two 16x16 accum tiles
    bf16x16 Bv0 = load_B_kcontig(Vt, N_NODES, h * HDIM,      key0);
    bf16x16 Bv1 = load_B_kcontig(Vt, N_NODES, h * HDIM + 16, key0);
    acc0 = wmma_bf16(Ap, Bv0, acc0);
    acc1 = wmma_bf16(Ap, Bv1, acc1);
  }

#pragma unroll
  for (int i = 0; i < 8; ++i) {
    int node = row0 + i + hs * 8;
    float inv = 1.0f / l_[i];
    attended[(size_t)node * DMODEL + h * HDIM + nn]      = acc0[i] * inv;
    attended[(size_t)node * DMODEL + h * HDIM + 16 + nn] = acc1[i] * inv;
  }
}

// ---------------------------------------------------------------------------
// 5) output projection + bias + residual (f32 epilogue)
// ---------------------------------------------------------------------------
__global__ void out_gemm(const __bf16* attb, const __bf16* WoT, const float* bo,
                         const float* x, float* hbuf) {
  int wid = blockIdx.x * (blockDim.x >> 5) + (threadIdx.x >> 5);
  int row0 = (wid >> 4) * 16;
  int col0 = (wid & 15) * 16;
  f32x8 acc = f32x8_zero();
#pragma unroll
  for (int k0 = 0; k0 < DMODEL; k0 += 32) {
    bf16x16 A = load_A16x32(attb, DMODEL, row0, k0);
    bf16x16 B = load_B_kcontig(WoT, DMODEL, col0, k0);
    acc = wmma_bf16(A, B, acc);
  }
  int lane = threadIdx.x & 31;
  int n = lane & 15, hs = lane >> 4;
  float bv = bo[col0 + n];
#pragma unroll
  for (int i = 0; i < 8; ++i) {
    int m = row0 + i + hs * 8;
    size_t idx = (size_t)m * DMODEL + col0 + n;
    hbuf[idx] = acc[i] + bv + x[idx];
  }
}

// ---------------------------------------------------------------------------
// 6) LayerNorm (one 256-thread block per row)
// ---------------------------------------------------------------------------
__global__ void layernorm(const float* hbuf, const float* w, const float* b,
                          float* out) {
  __shared__ float ws[8], ws2[8];
  int row = blockIdx.x, t = threadIdx.x;
  float v = hbuf[(size_t)row * DMODEL + t];
  float s = v, s2 = v * v;
#pragma unroll
  for (int off = 16; off >= 1; off >>= 1) {
    s  += __shfl_xor(s,  off, 32);
    s2 += __shfl_xor(s2, off, 32);
  }
  if ((t & 31) == 0) { ws[t >> 5] = s; ws2[t >> 5] = s2; }
  __syncthreads();
  float tot = 0.f, tot2 = 0.f;
#pragma unroll
  for (int j = 0; j < 8; ++j) { tot += ws[j]; tot2 += ws2[j]; }
  float mu  = tot * (1.0f / DMODEL);
  float var = tot2 * (1.0f / DMODEL) - mu * mu;
  out[(size_t)row * DMODEL + t] = (v - mu) * rsqrtf(var + 1e-5f) * w[t] + b[t];
}

// ---------------------------------------------------------------------------
extern "C" void kernel_launch(void* const* d_in, const int* in_sizes, int n_in,
                              void* d_out, int out_size, void* d_ws, size_t ws_size,
                              hipStream_t stream) {
  const float* x  = (const float*)d_in[0];
  const int*   ei = (const int*)d_in[1];
  const float* Wq = (const float*)d_in[2];
  const float* bq = (const float*)d_in[3];
  const float* Wk = (const float*)d_in[4];
  const float* bk = (const float*)d_in[5];
  const float* Wv = (const float*)d_in[6];
  const float* bv = (const float*)d_in[7];
  const float* Wo = (const float*)d_in[8];
  const float* bo = (const float*)d_in[9];
  const float* lw = (const float*)d_in[10];
  const float* lb = (const float*)d_in[11];
  float* out = (float*)d_out;
  int E = in_sizes[1] / 2;

  // workspace layout (all offsets 256B-aligned)
  char* ws = (char*)d_ws;
  uint32_t* adj  = (uint32_t*)(ws + 0);                       // 2 MB
  uint32_t* mask = (uint32_t*)(ws + (2u << 20));              // 2 MB
  __bf16*  xb    = (__bf16*) (ws + (4u << 20));               // 2 MB
  __bf16*  WqT   = (__bf16*) (ws + (6u << 20));               // 128 KB each
  __bf16*  WkT   = (__bf16*) (ws + (6u << 20) + 131072);
  __bf16*  WvT   = (__bf16*) (ws + (6u << 20) + 262144);
  __bf16*  WoT   = (__bf16*) (ws + (6u << 20) + 393216);
  __bf16*  Qb    = (__bf16*) (ws + (6u << 20) + 524288);      // 2 MB
  __bf16*  Kb    = (__bf16*) (ws + (8u << 20) + 524288);      // 2 MB
  __bf16*  Vt    = (__bf16*) (ws + (10u << 20) + 524288);     // 2 MB (V^T)
  float*   attended = (float*)(ws + (12u << 20) + 524288);    // 4 MB
  __bf16*  attb  = (__bf16*) (ws + (16u << 20) + 524288);     // 2 MB
  float*   hbuf  = (float*)  (ws + (18u << 20) + 524288);     // 4 MB

  // mask pipeline
  zero_words<<<512, 256, 0, stream>>>(adj, N_NODES * MASK_W);
  build_adj<<<(E + N_NODES + 255) / 256, 256, 0, stream>>>(ei, E, adj);
  reach2<<<N_NODES, 128, 0, stream>>>(adj, mask);

  // bf16 copies: activations plain, weights transposed
  cvt_bf16<<<1024, 256, 0, stream>>>(x, xb, N_NODES * DMODEL);
  cvt_bf16_T<<<256, 256, 0, stream>>>(Wq, WqT, DMODEL, DMODEL);
  cvt_bf16_T<<<256, 256, 0, stream>>>(Wk, WkT, DMODEL, DMODEL);
  cvt_bf16_T<<<256, 256, 0, stream>>>(Wv, WvT, DMODEL, DMODEL);
  cvt_bf16_T<<<256, 256, 0, stream>>>(Wo, WoT, DMODEL, DMODEL);

  // QKV: 3 * 4096 tiles, 8 waves per block
  qkv_gemm<<<1536, 256, 0, stream>>>(xb, WqT, WkT, WvT, bq, bk, bv, Qb, Kb, Vt);

  // flash attention: 2048 waves
  attn_flash<<<256, 256, 0, stream>>>(Qb, Kb, Vt, mask, attended);

  // output projection + residual, then layernorm
  cvt_bf16<<<1024, 256, 0, stream>>>(attended, attb, N_NODES * DMODEL);
  out_gemm<<<512, 256, 0, stream>>>(attb, WoT, bo, x, hbuf);
  layernorm<<<N_NODES, 256, 0, stream>>>(hbuf, lw, lb, out);
}
